// WindowMSA_39393440039334
// MI455X (gfx1250) — matrix operand order repaired and checked by reading, not compiled, not run on hardware
//
#include <hip/hip_runtime.h>

typedef __attribute__((ext_vector_type(16))) _Float16 v16h;
typedef __attribute__((ext_vector_type(8)))  _Float16 v8h;
typedef __attribute__((ext_vector_type(8)))  float    v8f;
typedef int v4i __attribute__((vector_size(16)));

#define WSZ 7
#define NH 3
#define CH 96
#define HD 32
#define WS2 49
#define HTOK 224
#define LTOK (224 * 224)

#if __has_builtin(__builtin_amdgcn_global_load_async_to_lds_b128) && \
    __has_builtin(__builtin_amdgcn_s_wait_asynccnt)
#define USE_ASYNC_LDS 1
#else
#define USE_ASYNC_LDS 0
#endif

// ---- WMMA fragment loaders (CDNA5 16-bit layouts, ISA 7.12.2) ----

// A matrix 16x32 (MxK), row-major source, leading dim `ld` (halves).
__device__ __forceinline__ v16h ld_afrag(const _Float16* base, int ld, int row, int k0) {
  const int lane = threadIdx.x & 31;
  const _Float16* p = base + (row + (lane & 15)) * ld + k0 + ((lane >> 4) << 3);
  v8h lo = *(const v8h*)p;
  v8h hi = *(const v8h*)(p + 16);
  v16h r;
#pragma unroll
  for (int i = 0; i < 8; ++i) { r[i] = lo[i]; r[i + 8] = hi[i]; }
  return r;
}

// B matrix 32x16 (KxN) from TRANSPOSED storage bt[n][k], ld in halves.
__device__ __forceinline__ v16h ld_bfrag(const _Float16* bt, int ld, int col0, int k0) {
  const int lane = threadIdx.x & 31;
  const _Float16* p = bt + (col0 + (lane & 15)) * ld + k0 + ((lane >> 4) << 3);
  v8h lo = *(const v8h*)p;
  v8h hi = *(const v8h*)(p + 16);
  v16h r;
#pragma unroll
  for (int i = 0; i < 8; ++i) { r[i] = lo[i]; r[i + 8] = hi[i]; }
  return r;
}

__device__ __forceinline__ v8f wmma_f16(v16h a, v16h b, v8f c) {
  return __builtin_amdgcn_wmma_f32_16x16x32_f16(false, a, false, b, (short)0, c,
                                                false, false);
}

// Compile-time-specialized projection: removes all runtime branching from the
// epilogue (MAT==0: Q with folded 1/sqrt(hd) scale; MAT==1: K; MAT==2: V transposed).
template <int MAT>
__device__ __forceinline__ void proj_phase(const _Float16* __restrict__ sXp,
                                           const _Float16* __restrict__ wt,
                                           const float* __restrict__ bias,
                                           _Float16* __restrict__ dst,
                                           int wave, int g, int ln) {
  for (int job = wave; job < 24; job += 8) {
    int mt = job / 6;
    int nt = job - mt * 6;
    v8f acc = {0.f, 0.f, 0.f, 0.f, 0.f, 0.f, 0.f, 0.f};
#pragma unroll
    for (int ks = 0; ks < 3; ++ks) {
      v16h a = ld_afrag(sXp, CH, mt * 16, ks * 32);
      v16h b = ld_bfrag(wt, CH, nt * 16, ks * 32);
      acc = wmma_f16(a, b, acc);
    }
    int col = nt * 16 + ln;
    float bv = bias[col];
#pragma unroll
    for (int r = 0; r < 8; ++r) {
      int row = mt * 16 + r + 8 * g;
      float v = acc[r] + bv;
      if (MAT == 0)      dst[row * CH + col] = (_Float16)(v * 0.17677669529663687f);
      else if (MAT == 1) dst[row * CH + col] = (_Float16)v;
      else               dst[col * 64 + row] = (_Float16)v;  // V transposed
    }
  }
}

// ---- prep: f32 weights -> f16 transposed in ws; gather padded rel-pos bias ----
__global__ void prep_kernel(const float* __restrict__ Wq, const float* __restrict__ Wk,
                            const float* __restrict__ Wv, const float* __restrict__ Wo,
                            const float* __restrict__ table, const int* __restrict__ relidx,
                            _Float16* __restrict__ wsW, float* __restrict__ biasPad) {
  int i = blockIdx.x * blockDim.x + threadIdx.x;
  if (i < 4 * CH * CH) {
    int m = i / (CH * CH);
    int r = i - m * (CH * CH);
    int nn = r / CH, kk = r - (r / CH) * CH;
    const float* W = (m == 0) ? Wq : (m == 1) ? Wk : (m == 2) ? Wv : Wo;
    wsW[i] = (_Float16)W[kk * CH + nn];  // transposed: wsW[m][n][k]
  }
  int j = i - 4 * CH * CH;
  if (j >= 0 && j < NH * 64 * 64) {
    int h = j / 4096;
    int rr = j - h * 4096;
    int q = rr >> 6, k = rr & 63;
    float v = 0.f;
    if (q < WS2 && k < WS2) v = table[h * 169 + relidx[q * WS2 + k]];
    biasPad[j] = v;
  }
}

// ---- fused window-MSA: 1 block = 1 window (49 tokens, padded to 64) ----
__global__ __launch_bounds__(256) void winmsa_kernel(
    const float* __restrict__ x, const float* __restrict__ x0,
    const float* __restrict__ bq, const float* __restrict__ bk,
    const float* __restrict__ bv, const float* __restrict__ bo,
    const _Float16* __restrict__ wsW, const float* __restrict__ biasPad,
    float* __restrict__ out) {
  // 56 KB LDS, manually laid out so the f32 async staging buffer (24 KB) can
  // overlay sQ+sK (dead until after the f16 conversion).
  __shared__ __align__(16) unsigned char smem[57344];
  _Float16* sX  = (_Float16*)(smem);          // 12288 B: X, later reused for O
  _Float16* sQ  = (_Float16*)(smem + 12288);  // 12288 B
  _Float16* sK  = (_Float16*)(smem + 24576);  // 12288 B
  _Float16* sVt = (_Float16*)(smem + 36864);  // 12288 B: V transposed [ch][tok]
  _Float16* sP  = (_Float16*)(smem + 49152);  //  8192 B: probs (one head)

  const int tid = threadIdx.x;
  const int wave = tid >> 5;
  const int lane = tid & 31;
  const int g = lane >> 4;
  const int ln = lane & 15;

  const int blk = blockIdx.x;
  const int n = blk >> 10;  // 1024 windows per image
  const int win = blk & 1023;
  const int by = win >> 5;
  const int bx = win & 31;
  const int tokBase = n * LTOK;

  // ---- Phase 0: window-gather X (f32) and convert to f16, zero-pad rows 49..63
#if USE_ASYNC_LDS
  {
    float* stage = (float*)(smem + 12288);  // 49*96 f32 = 18816 B (fits 24576)
    // 7 window rows, each a contiguous 7*96*4 = 2688 B strip -> 168 x 16 B chunks
    for (int c = tid; c < 7 * 168; c += 256) {
      int yr = c / 168;
      int off = (c - yr * 168) * 4;  // in floats
      const float* gsrc = x + (tokBase + (by * WSZ + yr) * HTOK + bx * WSZ) * CH + off;
      float* ldst = stage + yr * (WSZ * CH) + off;
      __builtin_amdgcn_global_load_async_to_lds_b128(
          (__attribute__((address_space(1))) v4i*)gsrc,
          (__attribute__((address_space(3))) v4i*)ldst, 0, 0);
    }
    __builtin_amdgcn_s_wait_asynccnt(0);
    __syncthreads();
    for (int idx = tid; idx < 64 * CH; idx += 256) {
      int row = idx / CH, ch = idx - row * CH;
      float v = (row < WS2) ? stage[row * CH + ch] : 0.f;
      sX[idx] = (_Float16)v;
    }
  }
#else
  for (int idx = tid; idx < 64 * CH; idx += 256) {
    int row = idx / CH, ch = idx - row * CH;
    float v = 0.f;
    if (row < WS2) {
      int rr = row / WSZ, cc = row - rr * WSZ;
      int l = tokBase + (by * WSZ + rr) * HTOK + bx * WSZ + cc;
      v = x[l * CH + ch];
    }
    sX[idx] = (_Float16)v;
  }
#endif
  __syncthreads();

  // ---- Phase 1: Q/K/V projections (branch-free, compile-time specialized)
  proj_phase<0>(sX, wsW,               bq, sQ,  wave, g, ln);
  proj_phase<1>(sX, wsW + CH * CH,     bk, sK,  wave, g, ln);
  proj_phase<2>(sX, wsW + 2 * CH * CH, bv, sVt, wave, g, ln);
  __syncthreads();

  // ---- Phase 2: attention, one head at a time (sP reused per head)
  for (int h = 0; h < NH; ++h) {
    if (wave < 4) {
      // scores: Q_h (64x32) @ K_h^T (32x64); sK row-major == Bt for K^T
      const int mt = wave;
      v16h a = ld_afrag(sQ, CH, mt * 16, h * HD);
      v8f s[4];
#pragma unroll
      for (int nt = 0; nt < 4; ++nt) {
        v8f z = {0.f, 0.f, 0.f, 0.f, 0.f, 0.f, 0.f, 0.f};
        s[nt] = wmma_f16(a, ld_bfrag(sK, CH, nt * 16, h * HD), z);
      }
      const float* bp = biasPad + h * 4096 + (mt * 16 + 8 * g) * 64 + ln;
#pragma unroll
      for (int nt = 0; nt < 4; ++nt)
#pragma unroll
        for (int r = 0; r < 8; ++r) s[nt][r] += bp[r * 64 + nt * 16];
#pragma unroll
      for (int nt = 0; nt < 4; ++nt)
        if (nt * 16 + ln >= WS2)
#pragma unroll
          for (int r = 0; r < 8; ++r) s[nt][r] = -1e30f;
      // row softmax: reduce across 16 lanes of each half-wave (same row group)
#pragma unroll
      for (int r = 0; r < 8; ++r) {
        float mx = fmaxf(fmaxf(s[0][r], s[1][r]), fmaxf(s[2][r], s[3][r]));
#pragma unroll
        for (int off = 1; off < 16; off <<= 1) mx = fmaxf(mx, __shfl_xor(mx, off, 32));
        float e[4];
        float sum = 0.f;
#pragma unroll
        for (int nt = 0; nt < 4; ++nt) {
          float ev = (nt * 16 + ln < WS2) ? __expf(s[nt][r] - mx) : 0.f;
          e[nt] = ev;
          sum += ev;
        }
#pragma unroll
        for (int off = 1; off < 16; off <<= 1) sum += __shfl_xor(sum, off, 32);
        float inv = 1.f / sum;
        int row = mt * 16 + r + 8 * g;
#pragma unroll
        for (int nt = 0; nt < 4; ++nt)
          sP[row * 64 + nt * 16 + ln] = (_Float16)(e[nt] * inv);
      }
    }
    __syncthreads();  // probs ready

    // O_h = P (64x64) @ V_h (64x32): 4 Mtiles * 2 Ntiles = 8 jobs, 1 per wave
    {
      const int mt = wave >> 1;
      const int col0 = h * HD + (wave & 1) * 16;
      v8f acc = {0.f, 0.f, 0.f, 0.f, 0.f, 0.f, 0.f, 0.f};
#pragma unroll
      for (int ks = 0; ks < 2; ++ks) {
        v16h a = ld_afrag(sP, 64, mt * 16, ks * 32);
        v16h b = ld_bfrag(sVt, 64, col0, ks * 32);
        acc = wmma_f16(a, b, acc);
      }
      __syncthreads();  // sP fully consumed before next head overwrites it
#pragma unroll
      for (int r = 0; r < 8; ++r) {
        int row = mt * 16 + r + 8 * g;
        sX[row * CH + col0 + ln] = (_Float16)acc[r];  // O overwrites X buffer
      }
    }
  }
  __syncthreads();

  // ---- Phase 3: out = O @ Wo + bo + x0, scattered back through window layout
  for (int job = wave; job < 24; job += 8) {
    int mt = job / 6;
    int nt = job - mt * 6;
    v8f acc = {0.f, 0.f, 0.f, 0.f, 0.f, 0.f, 0.f, 0.f};
    const _Float16* wt = wsW + 3 * (CH * CH);
#pragma unroll
    for (int ks = 0; ks < 3; ++ks) {
      v16h a = ld_afrag(sX, CH, mt * 16, ks * 32);
      v16h b = ld_bfrag(wt, CH, nt * 16, ks * 32);
      acc = wmma_f16(a, b, acc);
    }
    int col = nt * 16 + ln;
    float bias = bo[col];
#pragma unroll
    for (int r = 0; r < 8; ++r) {
      int m = mt * 16 + r + 8 * g;
      if (m < WS2) {
        int rr = m / WSZ, cc = m - rr * WSZ;
        int l = tokBase + (by * WSZ + rr) * HTOK + bx * WSZ + cc;
        int o = l * CH + col;
        out[o] = acc[r] + bias + x0[o];
      }
    }
  }
}

extern "C" void kernel_launch(void* const* d_in, const int* in_sizes, int n_in,
                              void* d_out, int out_size, void* d_ws, size_t ws_size,
                              hipStream_t stream) {
  const float* x     = (const float*)d_in[0];
  const float* x0    = (const float*)d_in[1];
  const float* table = (const float*)d_in[2];
  const int*   relid = (const int*)d_in[3];
  const float* Wq    = (const float*)d_in[4];
  const float* bq    = (const float*)d_in[5];
  const float* Wk    = (const float*)d_in[6];
  const float* bk    = (const float*)d_in[7];
  const float* Wv    = (const float*)d_in[8];
  const float* bv    = (const float*)d_in[9];
  const float* Wo    = (const float*)d_in[10];
  const float* bo    = (const float*)d_in[11];

  _Float16* wsW   = (_Float16*)d_ws;                                        // 73,728 B
  float*    biasP = (float*)((char*)d_ws + 4 * CH * CH * sizeof(_Float16)); // 49,152 B

  const int prepElems = 4 * CH * CH + NH * 64 * 64;  // 49,152
  prep_kernel<<<(prepElems + 255) / 256, 256, 0, stream>>>(Wq, Wk, Wv, Wo, table,
                                                           relid, wsW, biasP);
  winmsa_kernel<<<8192, 256, 0, stream>>>(x, x0, bq, bk, bv, bo, wsW, biasP,
                                          (float*)d_out);
}
